// PerformerAttention_17377437679787
// MI455X (gfx1250) — matrix operand order repaired and checked by reading, not compiled, not run on hardware
//
#include <hip/hip_runtime.h>
#include <math.h>

// Problem constants
#define T_SEQ 8192
#define B_SZ  4
#define C_DIM 1024
#define H_NUM 16

typedef __attribute__((ext_vector_type(16))) __bf16 bf16x16;
typedef __attribute__((ext_vector_type(8)))  float  f32x8;
typedef __attribute__((ext_vector_type(4)))  float  f32x4;
typedef __attribute__((ext_vector_type(4)))  unsigned int u32x4;
typedef __attribute__((ext_vector_type(8)))  int    i32x8;
typedef __attribute__((ext_vector_type(4)))  int    i32x4;

union AB16 { bf16x16 v; f32x4 f4[2]; __bf16 h[16]; };
union PK8  { __bf16 h[8]; f32x4 f4; };

static __device__ __forceinline__ f32x8 wmma_bf16(const AB16& a, const AB16& b, f32x8 c) {
  // v_wmma_f32_16x16x32_bf16: (neg_a, A, neg_b, B, c_mod, C, reuse_a, reuse_b)
  return __builtin_amdgcn_wmma_f32_16x16x32_bf16(false, a.v, false, b.v, (short)0, c, false, false);
}

static __device__ __forceinline__ f32x8 zero8() {
  f32x8 z = {0.f,0.f,0.f,0.f,0.f,0.f,0.f,0.f};
  return z;
}

// LDS aperture: flat address low 32 bits are the LDS byte offset (ISA 10.2).
static __device__ __forceinline__ unsigned lds_addr_of(const void* p) {
  return (unsigned)(unsigned long long)(size_t)p;
}

// ---------------------------------------------------------------------------
// TDM: DMA a 2D bf16 tile of 32 (contiguous, K) x 128 (rows) from global into
// LDS, padding each 32-elem (16-DWORD) row with 4 DWORDs -> 40-elem LDS rows.
// D# layout per CDNA5 ISA ch.8 (group0: count/lds/global/type; group1: dims).
// Toolchain: clang-23 therock lane -> 6-arg builtin (extra int32x8, zeroed).
static __device__ __forceinline__ void tdm_load_tile_32x128(unsigned lds_byte_addr,
                                                            const __bf16* gptr,
                                                            unsigned row_stride_elems) {
  unsigned long long ga = (unsigned long long)(size_t)(const void*)gptr;
  u32x4 g0 = {
    1u,                                               // count=1, user desc
    lds_byte_addr,                                    // lds_addr
    (unsigned)ga,                                     // global_addr[31:0]
    (unsigned)((ga >> 32) & 0x01ffffffu) | (2u << 30) // global_addr[56:32] | type=2
  };
  i32x8 g1 = {
    (int)((1u << 16) |            // data_size = 2 bytes
          (1u << 20) |            // pad_enable
          (3u << 22) |            // pad_interval: 16 DWORDs (row = 32 bf16)
          (3u << 25)),            // pad_amount: 4 DWORDs (8 bf16) -> stride 40
    (int)(32u << 16),             // tensor_dim0[15:0]=32 (== tile_dim0, no OOB)
    (int)(128u << 16),            // tensor_dim1[15:0]=128
    (int)(32u << 16),             // tile_dim0 = 32
    (int)128,                     // tile_dim1 = 128, tile_dim2 = 0
    (int)row_stride_elems,        // tensor_dim0_stride[31:0]
    0, 0
  };
  i32x4 g2 = {1, 1, 0, 0};        // tensor_dim2=1, tensor_dim3=1 (unused)
  i32x4 g3 = {0, 0, 0, 0};
  i32x8 g4 = {0, 0, 0, 0, 0, 0, 0, 0};   // extra operand on clang-23 lane
  __builtin_amdgcn_tensor_load_to_lds(g0, g1, g2, g3, g4, 0);
}

// ---------------------------------------------------------------------------
// elementwise f32 -> bf16 (4 elems / thread)
__global__ __launch_bounds__(256) void convert_f32_bf16(const float* __restrict__ in,
                                                        __bf16* __restrict__ out, size_t n) {
  size_t i = ((size_t)blockIdx.x * 256 + threadIdx.x) * 4;
  if (i >= n) return;
  f32x4 v = *(const f32x4*)&in[i];
  out[i + 0] = (__bf16)v.x;
  out[i + 1] = (__bf16)v.y;
  out[i + 2] = (__bf16)v.z;
  out[i + 3] = (__bf16)v.w;
}

// transpose + convert: in[K][N] f32 -> out[N][K] bf16
__global__ __launch_bounds__(256) void transpose_f32_to_bf16(const float* __restrict__ in,
                                                             __bf16* __restrict__ out,
                                                             int K, int N) {
  size_t idx = (size_t)blockIdx.x * 256 + threadIdx.x;
  if (idx >= (size_t)K * N) return;
  int n = (int)(idx / K), k = (int)(idx % K);
  out[idx] = (__bf16)in[(size_t)k * N + n];
}

// ---------------------------------------------------------------------------
// Big GEMM: C[M][N] = A[M][K] (bf16 row-major) @ Bt[N][K]^T (bf16, N-major)
// Block tile 128x128, 8 waves, wave tile 64x32 (4 m-tiles x 2 n-tiles).
// TDM double-buffered LDS staging: wave 0 issues tensor_load_to_lds for tile
// k+1 while all waves run WMMA on tile k; s_wait_tensorcnt + barrier flips.
template <int OUTF32>
__global__ __launch_bounds__(256) void gemm_bf16(const __bf16* __restrict__ A,
                                                 const __bf16* __restrict__ Bt,
                                                 void* __restrict__ Cout,
                                                 int M, int N, int K) {
  __shared__ __align__(16) __bf16 As[2][128 * 40];
  __shared__ __align__(16) __bf16 Bs[2][128 * 40];
  const int tid  = threadIdx.x;
  const int lane = tid & 31, wave = tid >> 5;
  const int wm = wave >> 2, wn = wave & 3;          // 2 x 4 wave grid
  const int mblk = blockIdx.y * 128, nblk = blockIdx.x * 128;
  const int lr = lane & 15, lh = lane >> 4;
  const bool issuer = (wave == 0);

  f32x8 acc[4][2];
#pragma unroll
  for (int i = 0; i < 4; ++i)
#pragma unroll
    for (int j = 0; j < 2; ++j) acc[i][j] = zero8();

  const __bf16* Abase = A + (size_t)mblk * K;
  const __bf16* Bbase = Bt + (size_t)nblk * K;

  if (issuer) {
    tdm_load_tile_32x128(lds_addr_of(&As[0][0]), Abase, (unsigned)K);
    tdm_load_tile_32x128(lds_addr_of(&Bs[0][0]), Bbase, (unsigned)K);
    __builtin_amdgcn_s_wait_tensorcnt(0);
  }
  __syncthreads();

  for (int k0 = 0; k0 < K; k0 += 32) {
    const int cur = (k0 >> 5) & 1, nxt = cur ^ 1;
    const bool more = (k0 + 32) < K;
    if (issuer && more) {
      tdm_load_tile_32x128(lds_addr_of(&As[nxt][0]), Abase + k0 + 32, (unsigned)K);
      tdm_load_tile_32x128(lds_addr_of(&Bs[nxt][0]), Bbase + k0 + 32, (unsigned)K);
    }

    AB16 a[4], b[2];
#pragma unroll
    for (int i = 0; i < 4; ++i) {
      int r = wm * 64 + i * 16 + lr;
      a[i].f4[0] = *(const f32x4*)&As[cur][r * 40 + lh * 8];       // K {0..7}/{8..15}
      a[i].f4[1] = *(const f32x4*)&As[cur][r * 40 + 16 + lh * 8];  // K {16..23}/{24..31}
    }
#pragma unroll
    for (int j = 0; j < 2; ++j) {
      int r = wn * 32 + j * 16 + lr;
      b[j].f4[0] = *(const f32x4*)&Bs[cur][r * 40 + lh * 8];       // K {0..7}/{8..15}
      b[j].f4[1] = *(const f32x4*)&Bs[cur][r * 40 + 16 + lh * 8];  // K {16..23}/{24..31}
    }
#pragma unroll
    for (int i = 0; i < 4; ++i)
#pragma unroll
      for (int j = 0; j < 2; ++j) acc[i][j] = wmma_bf16(a[i], b[j], acc[i][j]);

    if (more) {
      if (issuer) __builtin_amdgcn_s_wait_tensorcnt(0);
      __syncthreads();
    }
  }

#pragma unroll
  for (int i = 0; i < 4; ++i) {
#pragma unroll
    for (int e = 0; e < 8; ++e) {
      int m = mblk + wm * 64 + i * 16 + e + 8 * lh;
#pragma unroll
      for (int j = 0; j < 2; ++j) {
        int n = nblk + wn * 32 + j * 16 + lr;
        if (OUTF32) ((float*)Cout)[(size_t)m * N + n] = acc[i][j][e];
        else        ((__bf16*)Cout)[(size_t)m * N + n] = (__bf16)acc[i][j][e];
      }
    }
  }
}

// ---------------------------------------------------------------------------
// phi kernel: phi = exp(q @ proj - 0.5*||q||^2) / sqrt(F) per head.
// A rows from qkv (row stride 3072, col base col_base + h*64), B = projT in LDS.
// grid: (T/256, B*H); 8 waves, each 32 rows x 64 cols.
__global__ __launch_bounds__(256) void phi_kernel(const __bf16* __restrict__ qkv,
                                                  const __bf16* __restrict__ projT,
                                                  __bf16* __restrict__ phi_out,
                                                  int col_base) {
  __shared__ __align__(16) __bf16 Ps[64 * 72];
  const int tid = threadIdx.x, lane = tid & 31, wave = tid >> 5;
  const int bh = blockIdx.y, b = bh >> 4, h = bh & 15;
  const int t0 = blockIdx.x * 256 + wave * 32;
  const int lr = lane & 15, lh = lane >> 4;

#pragma unroll
  for (int p = 0; p < 2; ++p) {
    int row = p * 32 + (tid >> 3), c = (tid & 7) * 8;
    *(f32x4*)&Ps[row * 72 + c] = *(const f32x4*)&projT[row * 64 + c];
  }
  __syncthreads();

  const __bf16* Abase = qkv + (size_t)b * T_SEQ * 3072 + col_base + h * 64;
  f32x8 acc[2][4];
#pragma unroll
  for (int i = 0; i < 2; ++i)
#pragma unroll
    for (int j = 0; j < 4; ++j) acc[i][j] = zero8();
  float nsq[2] = {0.f, 0.f};

#pragma unroll
  for (int ks = 0; ks < 2; ++ks) {
    AB16 a[2], bb[4];
#pragma unroll
    for (int i = 0; i < 2; ++i) {
      int t = t0 + i * 16 + lr;
      const __bf16* ap = Abase + (size_t)t * 3072 + ks * 32;
      a[i].f4[0] = *(const f32x4*)(ap + lh * 8);
      a[i].f4[1] = *(const f32x4*)(ap + 16 + lh * 8);
      float s = 0.f;
#pragma unroll
      for (int e = 0; e < 16; ++e) { float x = (float)a[i].h[e]; s += x * x; }
      nsq[i] += s;
    }
#pragma unroll
    for (int j = 0; j < 4; ++j) {
      int r = j * 16 + lr;
      bb[j].f4[0] = *(const f32x4*)&Ps[r * 72 + ks * 32 + lh * 8];
      bb[j].f4[1] = *(const f32x4*)&Ps[r * 72 + ks * 32 + 16 + lh * 8];
    }
#pragma unroll
    for (int i = 0; i < 2; ++i)
#pragma unroll
      for (int j = 0; j < 4; ++j) acc[i][j] = wmma_bf16(a[i], bb[j], acc[i][j]);
  }
  // lanes r and r+16 jointly hold row (r&15)'s 64 elements
#pragma unroll
  for (int i = 0; i < 2; ++i) nsq[i] += __shfl_xor(nsq[i], 16);

#pragma unroll
  for (int i = 0; i < 2; ++i) {
#pragma unroll
    for (int e = 0; e < 8; ++e) {
      float ns = __shfl(nsq[i], e + (lh << 3));   // row e+8*lh of this m-tile
      int m = t0 + i * 16 + e + 8 * lh;
#pragma unroll
      for (int j = 0; j < 4; ++j) {
        int f = j * 16 + lr;
        float v = __expf(acc[i][j][e] - 0.5f * ns) * 0.125f; // 1/sqrt(64)
        phi_out[((size_t)bh * T_SEQ + m) * 64 + f] = (__bf16)v;
      }
    }
  }
}

// ---------------------------------------------------------------------------
// per-head 64-wide transpose (bf16): out[bh][c][t] = src[t][c]
// mode 0: in = kprime laid out [bh*T + t][64]; mode 1: in = qkv, v slice.
__global__ __launch_bounds__(256) void transpose_head(const __bf16* __restrict__ in,
                                                      __bf16* __restrict__ out,
                                                      int istride, int mode) {
  __shared__ __align__(16) __bf16 tile[64 * 72];
  const int tid = threadIdx.x;
  const int bh = blockIdx.y, b = bh >> 4, h = bh & 15;
  const int t0 = blockIdx.x * 64;
  const __bf16* src = (mode == 0)
      ? in + (size_t)bh * T_SEQ * istride
      : in + (size_t)b * T_SEQ * istride + 2048 + h * 64;
#pragma unroll
  for (int p = 0; p < 2; ++p) {
    int r = p * 32 + (tid >> 3), c = (tid & 7) * 8;
    *(f32x4*)&tile[r * 72 + c] = *(const f32x4*)&src[(size_t)(t0 + r) * istride + c];
  }
  __syncthreads();
  __bf16* dst = out + (size_t)bh * 64 * T_SEQ;
#pragma unroll
  for (int p = 0; p < 2; ++p) {
    int c = p * 32 + (tid >> 3), r = (tid & 7) * 8;
    PK8 pk;
#pragma unroll
    for (int e = 0; e < 8; ++e) pk.h[e] = tile[(r + e) * 72 + c];
    *(f32x4*)&dst[(size_t)c * T_SEQ + t0 + r] = pk.f4;
  }
}

// ---------------------------------------------------------------------------
// kv kernel: kvT[bh][d][f] = sum_t k'[t][f] * v[t][d]; also k_sum[bh][f].
// One block per (b,h); 8 waves each reduce a 1024-long T slice; LDS f32 reduce.
__global__ __launch_bounds__(256) void kv_kernel(const __bf16* __restrict__ kT,
                                                 const __bf16* __restrict__ vT,
                                                 __bf16* __restrict__ kvT,
                                                 float* __restrict__ ksum) {
  __shared__ float red[64 * 65];
  __shared__ float ks_s[64];
  const int tid = threadIdx.x, lane = tid & 31, wave = tid >> 5;
  const int bh = blockIdx.x;
  const int lr = lane & 15, lh = lane >> 4;

  for (int i = tid; i < 64 * 65; i += 256) red[i] = 0.f;
  if (tid < 64) ks_s[tid] = 0.f;
  __syncthreads();

  const __bf16* Ab = kT + (size_t)bh * 64 * T_SEQ;
  const __bf16* Bb = vT + (size_t)bh * 64 * T_SEQ;

  f32x8 acc[4][4];
#pragma unroll
  for (int i = 0; i < 4; ++i)
#pragma unroll
    for (int j = 0; j < 4; ++j) acc[i][j] = zero8();
  float ksp[4] = {0.f, 0.f, 0.f, 0.f};

  const int tbase = wave * 1024;
  for (int kk = 0; kk < 1024; kk += 32) {
    int t0 = tbase + kk;
    AB16 a[4], bb[4];
#pragma unroll
    for (int i = 0; i < 4; ++i) {
      const __bf16* ap = Ab + (size_t)(i * 16 + lr) * T_SEQ + t0;
      a[i].f4[0] = *(const f32x4*)(ap + lh * 8);
      a[i].f4[1] = *(const f32x4*)(ap + 16 + lh * 8);
      float s = 0.f;
#pragma unroll
      for (int e = 0; e < 16; ++e) s += (float)a[i].h[e];
      ksp[i] += s;
    }
#pragma unroll
    for (int j = 0; j < 4; ++j) {
      const __bf16* bp = Bb + (size_t)(j * 16 + lr) * T_SEQ + t0;
      bb[j].f4[0] = *(const f32x4*)(bp + lh * 8);
      bb[j].f4[1] = *(const f32x4*)(bp + 16 + lh * 8);
    }
#pragma unroll
    for (int i = 0; i < 4; ++i)
#pragma unroll
      for (int j = 0; j < 4; ++j) acc[i][j] = wmma_bf16(a[i], bb[j], acc[i][j]);
  }

#pragma unroll
  for (int i = 0; i < 4; ++i) ksp[i] += __shfl_xor(ksp[i], 16);
  if (lane < 16) {
#pragma unroll
    for (int i = 0; i < 4; ++i) atomicAdd(&ks_s[i * 16 + lr], ksp[i]);
  }
#pragma unroll
  for (int i = 0; i < 4; ++i)
#pragma unroll
    for (int j = 0; j < 4; ++j)
#pragma unroll
      for (int e = 0; e < 8; ++e) {
        int f = i * 16 + e + 8 * lh;
        int d = j * 16 + lr;
        atomicAdd(&red[f * 65 + d], acc[i][j][e]);
      }
  __syncthreads();

  for (int idx = tid; idx < 4096; idx += 256) {
    int d = idx >> 6, f = idx & 63;
    kvT[(size_t)bh * 4096 + idx] = (__bf16)red[f * 65 + d];  // kvT[d][f]
  }
  if (tid < 64) ksum[bh * 64 + tid] = ks_s[tid];
}

// ---------------------------------------------------------------------------
// denom[bh][t] = 1 / (q'[t,:] . k_sum + 1e-6)
__global__ __launch_bounds__(256) void denom_kernel(const __bf16* __restrict__ qprime,
                                                    const float* __restrict__ ksum,
                                                    float* __restrict__ denom) {
  __shared__ float ks[64];
  const int bh = blockIdx.y;
  const int t = blockIdx.x * 256 + threadIdx.x;
  if (threadIdx.x < 64) ks[threadIdx.x] = ksum[bh * 64 + threadIdx.x];
  __syncthreads();
  const __bf16* qp = qprime + ((size_t)bh * T_SEQ + t) * 64;
  float s = 0.f;
#pragma unroll
  for (int f = 0; f < 64; ++f) s += (float)qp[f] * ks[f];
  denom[(size_t)bh * T_SEQ + t] = 1.f / (s + 1e-6f);
}

// ---------------------------------------------------------------------------
// attn[b][t][h*64+d] = denom[bh][t] * sum_f q'[t][f] * kv[f][d]
__global__ __launch_bounds__(256) void attnout_kernel(const __bf16* __restrict__ qprime,
                                                      const __bf16* __restrict__ kvT,
                                                      const float* __restrict__ denom,
                                                      __bf16* __restrict__ attn) {
  __shared__ __align__(16) __bf16 Bs[64 * 72];
  const int tid = threadIdx.x, lane = tid & 31, wave = tid >> 5;
  const int bh = blockIdx.y, b = bh >> 4, h = bh & 15;
  const int t0 = blockIdx.x * 256 + wave * 32;
  const int lr = lane & 15, lh = lane >> 4;

  const __bf16* kvb = kvT + (size_t)bh * 4096;
#pragma unroll
  for (int p = 0; p < 2; ++p) {
    int row = p * 32 + (tid >> 3), c = (tid & 7) * 8;
    *(f32x4*)&Bs[row * 72 + c] = *(const f32x4*)&kvb[row * 64 + c];
  }
  __syncthreads();

  const __bf16* Ab = qprime + (size_t)bh * T_SEQ * 64;
  f32x8 acc[2][4];
#pragma unroll
  for (int i = 0; i < 2; ++i)
#pragma unroll
    for (int j = 0; j < 4; ++j) acc[i][j] = zero8();

#pragma unroll
  for (int ks = 0; ks < 2; ++ks) {
    AB16 a[2], bb[4];
#pragma unroll
    for (int i = 0; i < 2; ++i) {
      int t = t0 + i * 16 + lr;
      const __bf16* ap = Ab + (size_t)t * 64 + ks * 32;
      a[i].f4[0] = *(const f32x4*)(ap + lh * 8);
      a[i].f4[1] = *(const f32x4*)(ap + 16 + lh * 8);
    }
#pragma unroll
    for (int j = 0; j < 4; ++j) {
      int r = j * 16 + lr;
      bb[j].f4[0] = *(const f32x4*)&Bs[r * 72 + ks * 32 + lh * 8];
      bb[j].f4[1] = *(const f32x4*)&Bs[r * 72 + ks * 32 + 16 + lh * 8];
    }
#pragma unroll
    for (int i = 0; i < 2; ++i)
#pragma unroll
      for (int j = 0; j < 4; ++j) acc[i][j] = wmma_bf16(a[i], bb[j], acc[i][j]);
  }

#pragma unroll
  for (int i = 0; i < 2; ++i) {
#pragma unroll
    for (int e = 0; e < 8; ++e) {
      int m = t0 + i * 16 + e + 8 * lh;
      float dn = denom[(size_t)bh * T_SEQ + m];
#pragma unroll
      for (int j = 0; j < 4; ++j) {
        int d = j * 16 + lr;
        attn[((size_t)(b * T_SEQ) + m) * 1024 + h * 64 + d] = (__bf16)(acc[i][j][e] * dn);
      }
    }
  }
}

// ---------------------------------------------------------------------------
extern "C" void kernel_launch(void* const* d_in, const int* in_sizes, int n_in,
                              void* d_out, int out_size, void* d_ws, size_t ws_size,
                              hipStream_t stream) {
  const float* x      = (const float*)d_in[0];  // (B,T,C)
  const float* W_attn = (const float*)d_in[1];  // (C, 3C)
  const float* W_proj = (const float*)d_in[2];  // (C, C)
  const float* proj   = (const float*)d_in[3];  // (D, F) = (64,64)
  float* out = (float*)d_out;

  char* ws = (char*)d_ws;
  size_t off = 0;
  auto alloc = [&](size_t bytes) -> void* {
    void* p = ws + off;
    off = (off + bytes + 255) & ~(size_t)255;
    return p;
  };
  __bf16* qkv    = (__bf16*)alloc((size_t)32768 * 3072 * 2); // 201 MB
  __bf16* xb     = (__bf16*)alloc((size_t)32768 * 1024 * 2); // 67 MB (reused as attn)
  __bf16* wattnT = (__bf16*)alloc((size_t)3072 * 1024 * 2);
  __bf16* wprojT = (__bf16*)alloc((size_t)1024 * 1024 * 2);
  __bf16* projT  = (__bf16*)alloc((size_t)64 * 64 * 2);
  __bf16* qprime = (__bf16*)alloc((size_t)64 * T_SEQ * 64 * 2); // (B*H,T,F)
  __bf16* kprime = (__bf16*)alloc((size_t)64 * T_SEQ * 64 * 2); // reused as vT
  __bf16* kT     = (__bf16*)alloc((size_t)64 * 64 * T_SEQ * 2); // (B*H,F,T)
  __bf16* kvT    = (__bf16*)alloc((size_t)64 * 64 * 64 * 2);    // (B*H,D,F)
  float*  ksum   = (float*)alloc((size_t)64 * 64 * 4);
  float*  denom  = (float*)alloc((size_t)64 * T_SEQ * 4);
  __bf16* vT     = kprime;
  __bf16* attn   = xb;

  // 1) precision conversion + weight transposes
  convert_f32_bf16<<<32768, 256, 0, stream>>>(x, xb, (size_t)32768 * 1024);
  transpose_f32_to_bf16<<<12288, 256, 0, stream>>>(W_attn, wattnT, 1024, 3072);
  transpose_f32_to_bf16<<<4096, 256, 0, stream>>>(W_proj, wprojT, 1024, 1024);
  transpose_f32_to_bf16<<<16, 256, 0, stream>>>(proj, projT, 64, 64);

  // 2) qkv = x @ W_attn   (M=32768, N=3072, K=1024)
  gemm_bf16<0><<<dim3(24, 256), 256, 0, stream>>>(xb, wattnT, qkv, 32768, 3072, 1024);

  // 3) q' and k'
  phi_kernel<<<dim3(32, 64), 256, 0, stream>>>(qkv, projT, qprime, 0);
  phi_kernel<<<dim3(32, 64), 256, 0, stream>>>(qkv, projT, kprime, 1024);

  // 4) per-head transposes: k' -> kT, v -> vT (vT aliases kprime, launched after kT)
  transpose_head<<<dim3(128, 64), 256, 0, stream>>>(kprime, kT, 64, 0);
  transpose_head<<<dim3(128, 64), 256, 0, stream>>>(qkv, vT, 3072, 1);

  // 5) kv = k'^T @ v and k_sum
  kv_kernel<<<64, 256, 0, stream>>>(kT, vT, kvT, ksum);

  // 6) denom
  denom_kernel<<<dim3(32, 64), 256, 0, stream>>>(qprime, ksum, denom);

  // 7) attn = (q' @ kv) * denom
  attnout_kernel<<<dim3(32, 64), 256, 0, stream>>>(qprime, kvT, denom, attn);

  // 8) out = attn @ W_proj  (M=32768, N=1024, K=1024) -> fp32 d_out
  gemm_bf16<1><<<dim3(8, 256), 256, 0, stream>>>(attn, wprojT, out, 32768, 1024, 1024);
}
